// GeniePathAggregator_73847667687538
// MI455X (gfx1250) — compile-verified, matrix operand order checked by reading, not run on hardware
//
#include <hip/hip_runtime.h>
#include <hip/hip_bf16.h>
#include <math.h>

// GeniePath aggregator, fully fused for MI455X (gfx1250, wave32, WMMA bf16).
// N=50000 nodes, K=32 neighbors, D=128 features.

#define D            128
#define KN           32
#define NODES_PER_WG 16
#define CHUNK_NODES  4
#define BLOCK        256          // 8 wave32 waves
#define LSTRIDE      136          // padded bf16 row stride (elements): 272B rows, 16B-aligned granules

typedef __bf16 bf16;
typedef bf16  bf16x8 __attribute__((ext_vector_type(8)));
typedef bf16  v16bf  __attribute__((ext_vector_type(16)));
typedef float v8f    __attribute__((ext_vector_type(8)));

// ---- dynamic LDS layout (bytes) --------------------------------------------
#define SZ_WBUF  (128 * LSTRIDE * 2)              // transposed weight, bf16
#define SZ_ABUF  (128 * LSTRIDE * 2)              // neighbor rows (one 4-node chunk), bf16
#define SZ_YBUF  (128 * LSTRIDE * 2)              // y tile for chunk, bf16
#define SZ_SELF  (NODES_PER_WG * LSTRIDE * 2)     // self rows / reused as h_tmp, bf16
#define SZ_PA    (NODES_PER_WG * LSTRIDE * 2)     // pooled as bf16 A-matrix
#define SZ_XBUF  (NODES_PER_WG * D * 4)           // X = self@W_s, f32
#define SZ_PBUF  (NODES_PER_WG * D * 4)           // pooled, f32
#define SZ_VBUF  (D * 4)
#define SZ_LOG   (CHUNK_NODES * KN * 2 * 4)       // 2 partial sums per row
#define SZ_ALPHA (CHUNK_NODES * KN * 4)
#define SZ_GBUF  (4 * NODES_PER_WG * D * 4)       // i,f,o,cgate, f32

#define OFF_WBUF  0
#define OFF_ABUF  (OFF_WBUF  + SZ_WBUF)
#define OFF_YBUF  (OFF_ABUF  + SZ_ABUF)
#define OFF_SELF  (OFF_YBUF  + SZ_YBUF)
#define OFF_PA    (OFF_SELF  + SZ_SELF)
#define OFF_XBUF  (OFF_PA    + SZ_PA)
#define OFF_PBUF  (OFF_XBUF  + SZ_XBUF)
#define OFF_VBUF  (OFF_PBUF  + SZ_PBUF)
#define OFF_LOG   (OFF_VBUF  + SZ_VBUF)
#define OFF_ALPHA (OFF_LOG   + SZ_LOG)
#define OFF_GBUF  (OFF_ALPHA + SZ_ALPHA)
#define SMEM_BYTES (OFF_GBUF + SZ_GBUF)           // ~160.5 KB < 320 KB WGP LDS

// ---- fast transcendentals: gfx1250 native v_tanh_f32 / v_rcp_f32 -----------
__device__ __forceinline__ float fast_tanh(float x) {
#if __has_builtin(__builtin_amdgcn_tanhf)
  return __builtin_amdgcn_tanhf(x);
#else
  return tanhf(x);
#endif
}
__device__ __forceinline__ float fast_sigmoid(float x) {
#if __has_builtin(__builtin_amdgcn_tanhf)
  return __builtin_amdgcn_tanhf(0.5f * x) * 0.5f + 0.5f;   // sigmoid(x) = 0.5*tanh(x/2)+0.5
#else
  return 1.f / (1.f + __expf(-x));
#endif
}
__device__ __forceinline__ float fast_rcp(float x) {
#if __has_builtin(__builtin_amdgcn_rcpf)
  return __builtin_amdgcn_rcpf(x);
#else
  return 1.f / x;
#endif
}

// ---- WMMA fragment helpers (CDNA5 16x16x32 bf16 layouts) -------------------
union FragU {
  struct { bf16x8 lo, hi; } h;
  v16bf v;
};

// A-matrix (16x32 bf16) fragment from a row-major padded LDS buffer.
// lane<16: row=lane,    K 0..7 (vgpr0-3) and K 16..23 (vgpr4-7)
// lane>=16: row=lane-16, K 8..15 and K 24..31
// => two 16-byte ds loads at (kBase + half*8) and (+16 elements).
// B-matrix (32x16) uses identical code on the *transposed* weight buffer.
__device__ __forceinline__ v16bf load_frag(const bf16* buf, int rowBase, int kBase) {
  const int lane = threadIdx.x & 31;
  const int half = lane >> 4;
  const int m    = lane & 15;
  const bf16* p  = buf + (rowBase + m) * LSTRIDE + kBase + half * 8;
  FragU f;
  f.h.lo = *(const bf16x8*)(p);
  f.h.hi = *(const bf16x8*)(p + 16);
  return f.v;
}

// C/D 16x16 f32: lane<16 -> M=vgpr, N=lane ; lane>=16 -> M=vgpr+8, N=lane-16
__device__ __forceinline__ void store_acc_bf16(bf16* buf, int rowBase, int colBase, v8f acc) {
  const int lane = threadIdx.x & 31;
  const int half = lane >> 4;
  const int n    = lane & 15;
#pragma unroll
  for (int i = 0; i < 8; ++i)
    buf[(rowBase + half * 8 + i) * LSTRIDE + colBase + n] = (bf16)acc[i];
}

// full K=128 reduction (4 x k32 WMMA) for one 16x16 tile
__device__ __forceinline__ v8f wmma_k128(const bf16* Abuf, int rowBase, const v16bf bw[4]) {
  v8f acc = {};
#pragma unroll
  for (int kc = 0; kc < 4; ++kc) {
    v16bf a = load_frag(Abuf, rowBase, kc * 32);
    acc = __builtin_amdgcn_wmma_f32_16x16x32_bf16(false, a, false, bw[kc],
                                                  (short)0, acc, false, false);
  }
  return acc;
}

// ---- staging helpers --------------------------------------------------------
// dst[e*LSTRIDE + k] = (bf16) W[k*D + e]   (transpose so B-frags read contiguous K)
__device__ __forceinline__ void stage_weightT(bf16* dst, const float* __restrict__ W) {
  const float4* W4 = (const float4*)W;
#pragma unroll
  for (int it = 0; it < (D * D / 4) / BLOCK; ++it) {
    int idx = it * BLOCK + threadIdx.x;
    int k   = idx >> 5;               // 32 float4 per row
    int e0  = (idx & 31) << 2;
    float4 w = W4[idx];
    dst[(e0 + 0) * LSTRIDE + k] = (bf16)w.x;
    dst[(e0 + 1) * LSTRIDE + k] = (bf16)w.y;
    dst[(e0 + 2) * LSTRIDE + k] = (bf16)w.z;
    dst[(e0 + 3) * LSTRIDE + k] = (bf16)w.w;
  }
}

// row-major f32 -> padded bf16 rows
__device__ __forceinline__ void stage_rows_bf16(bf16* dst, const float* __restrict__ src, int nrows) {
  const float4* S4 = (const float4*)src;
  const int total = nrows * (D / 4);
  for (int idx = threadIdx.x; idx < total; idx += BLOCK) {
    int r  = idx >> 5;
    int e0 = (idx & 31) << 2;
    float4 w = S4[idx];
    bf16* p = dst + r * LSTRIDE + e0;
    p[0] = (bf16)w.x; p[1] = (bf16)w.y; p[2] = (bf16)w.z; p[3] = (bf16)w.w;
  }
}

// ---- fused kernel -----------------------------------------------------------
__global__ void __launch_bounds__(BLOCK)
genie_fused(const float* __restrict__ self_vecs,
            const float* __restrict__ neigh_vecs,
            const float* __restrict__ c_in,
            const float* __restrict__ W_s, const float* __restrict__ W_d,
            const float* __restrict__ vvec,
            const float* __restrict__ W_t, const float* __restrict__ W_i,
            const float* __restrict__ W_f, const float* __restrict__ W_o,
            const float* __restrict__ W_c,
            float* __restrict__ out_h, float* __restrict__ out_c)
{
  extern __shared__ char smem[];
  bf16*  WBUF  = (bf16*) (smem + OFF_WBUF);
  bf16*  ABUF  = (bf16*) (smem + OFF_ABUF);
  bf16*  YBUF  = (bf16*) (smem + OFF_YBUF);
  bf16*  SELFB = (bf16*) (smem + OFF_SELF);
  bf16*  PA    = (bf16*) (smem + OFF_PA);
  float* XBUF  = (float*)(smem + OFF_XBUF);
  float* PBUF  = (float*)(smem + OFF_PBUF);
  float* VBUF  = (float*)(smem + OFF_VBUF);
  float* LOGB  = (float*)(smem + OFF_LOG);     // [row][2] partial sums
  float* ALPHA = (float*)(smem + OFF_ALPHA);
  float* GBUF  = (float*)(smem + OFF_GBUF);

  const int t     = threadIdx.x;
  const int wave  = t >> 5;              // 8 waves: each owns output-column tile [wave*16, wave*16+16)
  const int node0 = blockIdx.x * NODES_PER_WG;

  // ---- Phase 0: stage v, self rows, W_s^T; prefetch chunk 0 neighbors ----
  __builtin_prefetch(neigh_vecs + (size_t)node0 * KN * D + (size_t)t * 64, 0, 1);
  stage_rows_bf16(SELFB, self_vecs + (size_t)node0 * D, NODES_PER_WG);
  if (t < D) VBUF[t] = vvec[t];
  stage_weightT(WBUF, W_s);
  __syncthreads();

  // ---- Phase 1: X = self @ W_s (one 16x16 tile per wave, K=128) ----
  {
    v16bf bw[4];
#pragma unroll
    for (int kc = 0; kc < 4; ++kc) bw[kc] = load_frag(WBUF, wave * 16, kc * 32);
    v8f acc = wmma_k128(SELFB, 0, bw);
    const int lane = t & 31, half = lane >> 4, n = lane & 15;
#pragma unroll
    for (int i = 0; i < 8; ++i)
      XBUF[(half * 8 + i) * D + wave * 16 + n] = acc[i];
  }
  __syncthreads();

  // ---- Phase 2: y = neigh @ W_d, attention softmax, pooled ----
  stage_weightT(WBUF, W_d);
  __syncthreads();
  v16bf bWd[4];                           // W_d column-tile fragments live in VGPRs all loop long
#pragma unroll
  for (int kc = 0; kc < 4; ++kc) bWd[kc] = load_frag(WBUF, wave * 16, kc * 32);

  for (int ch = 0; ch < NODES_PER_WG / CHUNK_NODES; ++ch) {
    const size_t nbase = (size_t)(node0 + ch * CHUNK_NODES) * KN * D;
    stage_rows_bf16(ABUF, neigh_vecs + nbase, CHUNK_NODES * KN);   // 128 rows
    if (ch + 1 < NODES_PER_WG / CHUNK_NODES) {                      // pull next chunk toward L2/L0
      const float* nx = neigh_vecs + (size_t)(node0 + (ch + 1) * CHUNK_NODES) * KN * D;
      __builtin_prefetch(nx + (size_t)t * 64, 0, 1);                // 256 threads x 256B lines = 64 KB
    }
    __syncthreads();

    // y tiles: 8 row-tiles x (this wave's) 1 col-tile, 4 WMMAs each
#pragma unroll
    for (int r = 0; r < 8; ++r) {
      v8f acc = wmma_k128(ABUF, r * 16, bWd);
      store_acc_bf16(YBUF, r * 16, wave * 16, acc);
    }
    __syncthreads();

    // logits[m] = sum_e tanh(x[node,e] + y[m,e]) * v[e]
    // all 256 threads: 2 threads per row, 64 elements each (native v_tanh_f32)
    {
      const int m  = t >> 1;                  // row in chunk (node j = m>>5)
      const int hh = t & 1;
      const float* xr = XBUF + (ch * CHUNK_NODES + (m >> 5)) * D + hh * 64;
      const bf16*  yr = YBUF + m * LSTRIDE + hh * 64;
      const float* vr = VBUF + hh * 64;
      float s = 0.f;
#pragma unroll 4
      for (int e = 0; e < 64; ++e) s += fast_tanh(xr[e] + (float)yr[e]) * vr[e];
      LOGB[m * 2 + hh] = s;
    }
    __syncthreads();

    // softmax over K=32 (one wave per node; wave32 == K)
    if (wave < CHUNK_NODES) {
      const int lane = t & 31;
      const int row  = wave * KN + lane;
      float l = LOGB[row * 2] + LOGB[row * 2 + 1];
      float mx = l;
#pragma unroll
      for (int o = 16; o > 0; o >>= 1) mx = fmaxf(mx, __shfl_xor(mx, o, 32));
      float e = __expf(l - mx);
      float sm = e;
#pragma unroll
      for (int o = 16; o > 0; o >>= 1) sm += __shfl_xor(sm, o, 32);
      ALPHA[wave * KN + lane] = e * fast_rcp(sm);
    }
    __syncthreads();

    // pooled[node,e] = sum_k alpha[k] * y[k,e]
    {
      const int e  = t & 127;
      const int j0 = t >> 7;
#pragma unroll
      for (int jj = 0; jj < 2; ++jj) {
        const int j = j0 * 2 + jj;
        float s = 0.f;
#pragma unroll
        for (int k = 0; k < KN; ++k)
          s += ALPHA[j * KN + k] * (float)YBUF[(j * KN + k) * LSTRIDE + e];
        PBUF[(ch * CHUNK_NODES + j) * D + e] = s;
      }
    }
    __syncthreads();
  }

  // ---- Phase 3: h_tmp = tanh(pooled @ W_t) ----
  stage_weightT(WBUF, W_t);
  for (int idx = t; idx < NODES_PER_WG * D; idx += BLOCK) {
    const int r = idx >> 7, e = idx & 127;
    PA[r * LSTRIDE + e] = (bf16)PBUF[idx];
  }
  __syncthreads();
  {
    v16bf bw[4];
#pragma unroll
    for (int kc = 0; kc < 4; ++kc) bw[kc] = load_frag(WBUF, wave * 16, kc * 32);
    v8f acc = wmma_k128(PA, 0, bw);
    const int lane = t & 31, half = lane >> 4, n = lane & 15;
#pragma unroll
    for (int i = 0; i < 8; ++i)                               // reuse SELFB as bf16 h_tmp
      SELFB[(half * 8 + i) * LSTRIDE + wave * 16 + n] = (bf16)fast_tanh(acc[i]);
  }
  __syncthreads();

  // ---- Phase 4: four gate GEMMs, h_tmp A-fragments cached in VGPRs ----
  v16bf aH[4];
#pragma unroll
  for (int kc = 0; kc < 4; ++kc) aH[kc] = load_frag(SELFB, 0, kc * 32);
  const float* gw[4] = { W_i, W_f, W_o, W_c };
  for (int g = 0; g < 4; ++g) {
    stage_weightT(WBUF, gw[g]);
    __syncthreads();
    v8f acc = {};
#pragma unroll
    for (int kc = 0; kc < 4; ++kc)
      acc = __builtin_amdgcn_wmma_f32_16x16x32_bf16(false, aH[kc], false,
              load_frag(WBUF, wave * 16, kc * 32), (short)0, acc, false, false);
    const int lane = t & 31, half = lane >> 4, n = lane & 15;
    float* G = GBUF + g * (NODES_PER_WG * D);
#pragma unroll
    for (int i = 0; i < 8; ++i) {
      const float x = acc[i];
      G[(half * 8 + i) * D + wave * 16 + n] = (g == 3) ? fast_tanh(x) : fast_sigmoid(x);
    }
    __syncthreads();
  }

  // ---- Phase 5: c_new = f*c + i*cg ; h = o*tanh(c_new) ----
  {
    const float* GI = GBUF;
    const float* GF = GBUF + 1 * NODES_PER_WG * D;
    const float* GO = GBUF + 2 * NODES_PER_WG * D;
    const float* GC = GBUF + 3 * NODES_PER_WG * D;
    for (int idx = t; idx < NODES_PER_WG * D; idx += BLOCK) {
      const int r = idx >> 7, e = idx & 127;
      const size_t gidx = (size_t)(node0 + r) * D + e;
      const float co = c_in[gidx];
      const float cn = GF[idx] * co + GI[idx] * GC[idx];
      out_h[gidx] = GO[idx] * fast_tanh(cn);
      out_c[gidx] = cn;
    }
  }
}

// ---- host-side launcher -----------------------------------------------------
extern "C" void kernel_launch(void* const* d_in, const int* in_sizes, int n_in,
                              void* d_out, int out_size, void* d_ws, size_t ws_size,
                              hipStream_t stream) {
  const float* self  = (const float*)d_in[0];
  const float* neigh = (const float*)d_in[1];
  const float* c_in  = (const float*)d_in[2];
  const float* W_s   = (const float*)d_in[3];
  const float* W_d   = (const float*)d_in[4];
  const float* vvec  = (const float*)d_in[5];
  const float* W_t   = (const float*)d_in[6];
  const float* W_i   = (const float*)d_in[7];
  const float* W_f   = (const float*)d_in[8];
  const float* W_o   = (const float*)d_in[9];
  const float* W_c   = (const float*)d_in[10];

  const int N = in_sizes[0] / D;                 // 50000 (divisible by NODES_PER_WG)
  float* out_h = (float*)d_out;
  float* out_c = out_h + (size_t)N * D;

  // >64 KB dynamic LDS: opt in (deterministic; no stream ops, graph-capture safe)
  (void)hipFuncSetAttribute((const void*)genie_fused,
                            hipFuncAttributeMaxDynamicSharedMemorySize,
                            (int)SMEM_BYTES);

  dim3 grid(N / NODES_PER_WG);
  genie_fused<<<grid, BLOCK, SMEM_BYTES, stream>>>(
      self, neigh, c_in, W_s, W_d, vvec, W_t, W_i, W_f, W_o, W_c, out_h, out_c);
}
